// SymbolicAttention_19327352832446
// MI455X (gfx1250) — matrix operand order repaired and checked by reading, not compiled, not run on hardware
//
#include <hip/hip_runtime.h>
#include <hip/hip_bf16.h>
#include <math.h>

#define B_   2
#define T_   2048
#define H_   16
#define D_   64
#define C_   1024           // N_EMBD
#define M_   (B_ * T_)      // 4096 rows of x
#define N2_  (2 * C_)       // 2048 cols of W_attn

typedef __bf16 bf16_t;
typedef __attribute__((ext_vector_type(8)))  __bf16 bf8v;
typedef __attribute__((ext_vector_type(16))) __bf16 bf16v;
typedef __attribute__((ext_vector_type(8)))  float  f8v;
typedef __attribute__((ext_vector_type(4)))  unsigned int u32x4;
typedef __attribute__((ext_vector_type(8)))  unsigned int u32x8;

__device__ inline f8v zero_f8() {
    f8v z;
#pragma unroll
    for (int i = 0; i < 8; ++i) z[i] = 0.0f;
    return z;
}

__device__ inline bf16v cat16(bf8v lo, bf8v hi) {
    return __builtin_shufflevector(lo, hi, 0,1,2,3,4,5,6,7,8,9,10,11,12,13,14,15);
}

// A-fragment, 16x32 bf16 (M x K). src row-major with leading dim ld (elements).
// Lanes 0-15: row r0+lane, K = {kc+0..15}
// Lanes 16-31: row r0+lane-16, K = {kc+16..31}
__device__ inline bf16v load_a_frag(const bf16_t* src, int ld, int r0, int kc, int lane) {
    int half = lane >> 4, lm = lane & 15;
    const bf16_t* p = src + (size_t)(r0 + lm) * ld + kc + half * 16;
    bf8v lo = *(const bf8v*)p;
    bf8v hi = *(const bf8v*)(p + 8);
    return cat16(lo, hi);
}

// B-fragment, 32x16 bf16 (K x N), from K-major storage: element (k,n) at srcT[n*ld + k].
// Lane = n%16 + 16*(k>=16); each lane holds 16 contiguous K values.
__device__ inline bf16v load_b_frag(const bf16_t* srcT, int ld, int n0, int kc, int lane) {
    int half = lane >> 4, lm = lane & 15;
    const bf16_t* p = srcT + (size_t)(n0 + lm) * ld + kc + half * 16;
    bf8v lo = *(const bf8v*)p;
    bf8v hi = *(const bf8v*)(p + 8);
    return cat16(lo, hi);
}

__device__ inline f8v wmma_bf16(bf16v a, bf16v b, f8v c) {
    return __builtin_amdgcn_wmma_f32_16x16x32_bf16(false, a, false, b, (short)0, c,
                                                   false, false);
}

// TDM: DMA one 32(K-contig) x 64(N-rows) bf16 tile of Wt into LDS at lds_off.
// D# built per CDNA5 ISA (2D, groups 0+1 only).
// The LDS base pointer is passed as an (unused-in-template) operand so the
// tile escapes into the asm and the "memory" clobber covers the TDM's writes.
__device__ inline void tdm_load_wtile(const bf16_t* wt, int n_blk, int kc,
                                      unsigned lds_off, bf16_t* lds_base) {
    unsigned long long ga =
        (unsigned long long)(uintptr_t)wt +
        ((unsigned long long)n_blk * C_ + (unsigned long long)kc) * 2ull;
    u32x4 g0;                                 // group 0
    g0[0] = 1u;                               // count=1, user mode
    g0[1] = lds_off;                          // lds_addr (bytes)
    g0[2] = (unsigned)(ga & 0xffffffffull);   // global_addr[31:0]
    g0[3] = (unsigned)((ga >> 32) & 0x1ffffffull) | (2u << 30);  // [56:32] | type=2
    u32x8 g1;                                 // group 1
    g1[0] = 1u << 16;                         // data_size = 2B, no multicast
    g1[1] = ((unsigned)C_ & 0xffffu) << 16;   // tensor_dim0 = 1024 (k extent)
    g1[2] = ((unsigned)N2_ & 0xffffu) << 16;  // tensor_dim1 = 2048 (n rows)
    g1[3] = 32u << 16;                        // tile_dim0 = 32 (contig k)
    g1[4] = 64u;                              // tile_dim1 = 64, tile_dim2 = 0
    g1[5] = (unsigned)C_;                     // tensor_dim0_stride = 1024
    g1[6] = 0u;
    g1[7] = 0u;
    asm volatile("tensor_load_to_lds %0, %1"
                 :: "s"(g0), "s"(g1), "r"(lds_base) : "memory");
}

// ---------------------------------------------------------------- converters
__global__ __launch_bounds__(256) void cvt_x_kernel(const float* __restrict__ x,
                                                    bf16_t* __restrict__ xb) {
    int idx = blockIdx.x * 256 + threadIdx.x;
    if (idx < M_ * C_) xb[idx] = (bf16_t)x[idx];
}

// Wt[n][k] = W_attn[k][n], bf16 (K-major so B-fragments / TDM rows are contiguous)
__global__ __launch_bounds__(256) void cvt_w_kernel(const float* __restrict__ W,
                                                    bf16_t* __restrict__ wt) {
    int idx = blockIdx.x * 256 + threadIdx.x;
    if (idx >= N2_ * C_) return;
    int n = idx / C_, k = idx - n * C_;
    wt[(size_t)n * C_ + k] = (bf16_t)W[(size_t)k * N2_ + n];
}

// vT[(b*H+j)][d][t] = bf16( sum_i x[b,t,i*64+d] * v_tmp[i,j] )
__global__ __launch_bounds__(256) void vprep_kernel(const float* __restrict__ x,
                                                    const float* __restrict__ v_tmp,
                                                    bf16_t* __restrict__ vt) {
    int idx = blockIdx.x * 256 + threadIdx.x;
    if (idx >= B_ * H_ * D_ * T_) return;
    int t = idx & (T_ - 1);
    int d = (idx >> 11) & (D_ - 1);
    int j = (idx >> 17) & (H_ - 1);
    int b = idx >> 21;
    const float* xr = x + ((size_t)(b * T_ + t)) * C_ + d;
    float acc = 0.0f;
#pragma unroll
    for (int i = 0; i < H_; ++i) acc += xr[i * D_] * v_tmp[i * H_ + j];
    vt[idx] = (bf16_t)acc;
}

// ------------------------------------------------------- QK projection GEMM
// qk = x @ W_attn ; q/k scattered to [b*H+h][t][d] bf16 layouts.
// Weight tiles staged into double-buffered LDS by the Tensor Data Mover;
// TENSORcnt's in-order guarantee makes "wait <= 1" mean current tile ready.
__global__ __launch_bounds__(256) void gemm_qk_kernel(const bf16_t* __restrict__ xb,
                                                      const bf16_t* __restrict__ wt,
                                                      bf16_t* __restrict__ qb,
                                                      bf16_t* __restrict__ kb) {
    __shared__ bf16_t btile[2][64 * 32];      // 2 x 4 KB, LDS offsets 0 / 4096

    int lane = threadIdx.x & 31;
    int w    = threadIdx.x >> 5;              // 8 waves
    int m0   = blockIdx.x * 128 + (w & 3) * 32;
    int n_blk = blockIdx.y * 64;
    int n_loc = (w >> 2) * 32;                // wave's N offset inside the LDS tile

    // one-time init with real ds_stores (TDM overwrites after the barrier);
    // prevents "never-stored LDS" loads from being folded to undef.
    for (int i = threadIdx.x; i < 2 * 64 * 32; i += 256)
        (&btile[0][0])[i] = (bf16_t)0.0f;
    __syncthreads();

    if (threadIdx.x < 32)                     // pre-issue first tile
        tdm_load_wtile(wt, n_blk, 0, 0u, &btile[0][0]);

    f8v acc00 = zero_f8(), acc01 = zero_f8(), acc10 = zero_f8(), acc11 = zero_f8();

    int cur = 0;
#pragma unroll 2
    for (int kc = 0; kc < C_; kc += 32) {
        if (threadIdx.x < 32) {
            if (kc + 32 < C_) {               // prefetch next slab into other buffer
                tdm_load_wtile(wt, n_blk, kc + 32,
                               (unsigned)(cur ^ 1) * 4096u, &btile[0][0]);
                __builtin_amdgcn_s_wait_tensorcnt(1);  // current tile complete
            } else {
                __builtin_amdgcn_s_wait_tensorcnt(0);  // last tile complete
            }
        }
        __syncthreads();                      // tile visible to all 8 waves

        __builtin_prefetch(xb + (size_t)(m0 + (lane & 15)) * C_ + kc + 32, 0, 3);

        const bf16_t* bt = &btile[cur][0];
        bf16v a0 = load_a_frag(xb, C_, m0,      kc, lane);
        bf16v a1 = load_a_frag(xb, C_, m0 + 16, kc, lane);
        bf16v b0 = load_b_frag(bt, 32, n_loc,      0, lane);   // ds_read_b128
        bf16v b1 = load_b_frag(bt, 32, n_loc + 16, 0, lane);
        acc00 = wmma_bf16(a0, b0, acc00);
        acc01 = wmma_bf16(a0, b1, acc01);
        acc10 = wmma_bf16(a1, b0, acc10);
        acc11 = wmma_bf16(a1, b1, acc11);

        __syncthreads();                      // reads done before buffer reuse
        cur ^= 1;
    }

    int half = lane >> 4, lm = lane & 15;
    int n0 = n_blk + n_loc;
    f8v accs[2][2] = {{acc00, acc01}, {acc10, acc11}};
#pragma unroll
    for (int mf = 0; mf < 2; ++mf)
#pragma unroll
        for (int nf = 0; nf < 2; ++nf)
#pragma unroll
            for (int j = 0; j < 8; ++j) {
                int m = m0 + mf * 16 + j + half * 8;
                int n = n0 + nf * 16 + lm;
                bf16_t val = (bf16_t)accs[mf][nf][j];
                int b = m >> 11, t = m & (T_ - 1);
                if (n < C_) {
                    int h = n >> 6, d = n & 63;
                    qb[(((size_t)(b * H_ + h)) * T_ + t) * D_ + d] = val;
                } else {
                    int n2 = n - C_;
                    int h = n2 >> 6, d = n2 & 63;
                    kb[(((size_t)(b * H_ + h)) * T_ + t) * D_ + d] = val;
                }
            }
}

// ------------------------------------------------------ fused ALiBi attention
// One block = (b,h) x 128 queries; 8 waves x 16 query rows; flash-style online
// softmax, 32 keys per step (4 WMMA for S, 4 WMMA for P.V).
__global__ __launch_bounds__(256) void attn_kernel(const bf16_t* __restrict__ qb,
                                                   const bf16_t* __restrict__ kbuf,
                                                   const bf16_t* __restrict__ vt,
                                                   float* __restrict__ yw) {
    __shared__ bf16_t plds[8 * 16 * 32];      // per-wave 16x32 P staging (8 KB)

    int lane = threadIdx.x & 31;
    int w    = threadIdx.x >> 5;
    int bh   = blockIdx.x >> 4;               // b*H + h
    int qblk = blockIdx.x & 15;
    int b = bh >> 4, h = bh & 15;
    int q0 = qblk * 128 + w * 16;
    int half = lane >> 4, lm = lane & 15;

    const bf16_t* Q = qb   + (size_t)bh * T_ * D_;
    const bf16_t* K = kbuf + (size_t)bh * T_ * D_;
    const bf16_t* V = vt   + (size_t)bh * D_ * T_;   // [d][t]
    bf16_t* myP = plds + w * (16 * 32);

    const float scale = 0.125f;                       // 1/sqrt(64)
    float slope = exp2f(-0.5f * (float)(h + 1));      // ALiBi, H=16 pow2

    bf16v qa0 = load_a_frag(Q, D_, q0, 0,  lane);
    bf16v qa1 = load_a_frag(Q, D_, q0, 32, lane);

    f8v yacc[4];
    float rm[8], rs[8];
#pragma unroll
    for (int f = 0; f < 4; ++f) yacc[f] = zero_f8();
#pragma unroll
    for (int j = 0; j < 8; ++j) { rm[j] = -1e30f; rs[j] = 0.0f; }

#pragma unroll 2
    for (int kb0 = 0; kb0 < q0 + 16; kb0 += 32) {
        // ---- S = Q . K^T (two 16-key tiles, K-dim = 64 = 2 x 32)
        f8v s0 = zero_f8(), s1 = zero_f8();
        s0 = wmma_bf16(qa0, load_b_frag(K, D_, kb0,      0,  lane), s0);
        s0 = wmma_bf16(qa1, load_b_frag(K, D_, kb0,      32, lane), s0);
        s1 = wmma_bf16(qa0, load_b_frag(K, D_, kb0 + 16, 0,  lane), s1);
        s1 = wmma_bf16(qa1, load_b_frag(K, D_, kb0 + 16, 32, lane), s1);

        // ---- bias + causal mask + online softmax (per C-frag row)
#pragma unroll
        for (int j = 0; j < 8; ++j) {
            int qi  = q0 + j + half * 8;
            int k0i = kb0 + lm;
            int k1i = kb0 + 16 + lm;
            float v0 = s0[j] * scale + slope * (float)(k0i - qi);
            float v1 = s1[j] * scale + slope * (float)(k1i - qi);
            if (k0i > qi) v0 = -1e30f;
            if (k1i > qi) v1 = -1e30f;

            float mx = fmaxf(v0, v1);
#pragma unroll
            for (int off = 1; off < 16; off <<= 1)
                mx = fmaxf(mx, __shfl_xor(mx, off, 16));
            float nm   = fmaxf(rm[j], mx);
            float corr = expf(rm[j] - nm);
            float p0   = expf(v0 - nm);
            float p1   = expf(v1 - nm);
            float ps   = p0 + p1;
#pragma unroll
            for (int off = 1; off < 16; off <<= 1)
                ps += __shfl_xor(ps, off, 16);
            rm[j] = nm;
            rs[j] = rs[j] * corr + ps;
#pragma unroll
            for (int f = 0; f < 4; ++f) yacc[f][j] *= corr;

            int row = j + half * 8;
            myP[row * 32 + lm]      = (bf16_t)p0;     // same-wave LDS, in-order
            myP[row * 32 + 16 + lm] = (bf16_t)p1;
        }

        // ---- P (via LDS transpose to A-layout) . V
        bf16v pa = load_a_frag(myP, 32, 0, 0, lane);
#pragma unroll
        for (int f = 0; f < 4; ++f) {
            bf16v vb = load_b_frag(V, T_, f * 16, kb0, lane);  // n=d-chunk, K=keys
            yacc[f] = wmma_bf16(pa, vb, yacc[f]);
        }
    }

    // ---- normalize + store y[b][t][h][d] fp32
#pragma unroll
    for (int f = 0; f < 4; ++f)
#pragma unroll
        for (int j = 0; j < 8; ++j) {
            int qi = q0 + j + half * 8;
            int d  = f * 16 + lm;
            yw[(((size_t)b * T_ + qi) * H_ + h) * D_ + d] = yacc[f][j] / rs[j];
        }
}

// ------------------------------------------------------------- output mixing
__global__ __launch_bounds__(256) void proj_kernel(const float* __restrict__ yw,
                                                   const float* __restrict__ proj,
                                                   float* __restrict__ out) {
    int idx = blockIdx.x * 256 + threadIdx.x;
    if (idx >= B_ * T_ * C_) return;
    int c  = idx & (C_ - 1);
    int bt = idx >> 10;                       // b*T + t
    int i  = c >> 6, d = c & 63;
    const float* yr = yw + ((size_t)bt * H_) * D_ + d;
    float acc = 0.0f;
#pragma unroll
    for (int j = 0; j < H_; ++j) acc += yr[j * D_] * proj[i * H_ + j];
    out[idx] = acc;
}

// -------------------------------------------------------------------- launch
extern "C" void kernel_launch(void* const* d_in, const int* in_sizes, int n_in,
                              void* d_out, int out_size, void* d_ws, size_t ws_size,
                              hipStream_t stream) {
    (void)in_sizes; (void)n_in; (void)out_size; (void)ws_size;
    const float* x      = (const float*)d_in[0];
    const float* W_attn = (const float*)d_in[1];
    const float* v_tmp  = (const float*)d_in[2];
    const float* proj   = (const float*)d_in[3];
    float*       out    = (float*)d_out;

    char* ws = (char*)d_ws;
    const size_t SZ_XB = (size_t)M_ * C_ * 2;          // 8 MB
    const size_t SZ_WT = (size_t)N2_ * C_ * 2;         // 4 MB
    const size_t SZ_QB = (size_t)B_ * H_ * T_ * D_ * 2;// 8 MB
    const size_t SZ_KB = SZ_QB;
    const size_t SZ_VT = SZ_QB;

    bf16_t* xb = (bf16_t*)ws;                 ws += SZ_XB;
    bf16_t* wt = (bf16_t*)ws;                 ws += SZ_WT;
    bf16_t* qb = (bf16_t*)ws;                 ws += SZ_QB;
    bf16_t* kb = (bf16_t*)ws;                 ws += SZ_KB;
    bf16_t* vt = (bf16_t*)ws;                 ws += SZ_VT;
    float*  yw = (float*)ws;                  // 16 MB

    cvt_x_kernel<<<(M_ * C_ + 255) / 256, 256, 0, stream>>>(x, xb);
    cvt_w_kernel<<<(N2_ * C_ + 255) / 256, 256, 0, stream>>>(W_attn, wt);
    vprep_kernel<<<(B_ * H_ * D_ * T_ + 255) / 256, 256, 0, stream>>>(x, v_tmp, vt);
    gemm_qk_kernel<<<dim3(M_ / 128, N2_ / 64), 256, 0, stream>>>(xb, wt, qb, kb);
    attn_kernel<<<B_ * H_ * (T_ / 128), 256, 0, stream>>>(qb, kb, vt, yw);
    proj_kernel<<<(B_ * T_ * C_ + 255) / 256, 256, 0, stream>>>(yw, proj, out);
}
